// Mamba_53910429499975
// MI455X (gfx1250) — matrix operand order repaired and checked by reading, not compiled, not run on hardware
//
#include <hip/hip_runtime.h>
#include <hip/hip_bf16.h>
#include <math.h>

// ---------------- model constants ----------------
#define T_SEQ    2048
#define D_MODEL  1024
#define N_LAYER  2
#define VOCAB    50257
#define D_STATE  16
#define D_CONV   4
#define D_INNER  2048   // EXPAND * D_MODEL
#define DT_RANK  64
#define EPS_RMS  1e-5f

typedef __attribute__((ext_vector_type(16))) _Float16 v16h;
typedef __attribute__((ext_vector_type(8)))  _Float16 v8h;
typedef __attribute__((ext_vector_type(8)))  float    v8f;

// Assemble a v16h fragment from two contiguous 16B chunks `gap` halfs apart.
__device__ __forceinline__ v16h load_frag(const _Float16* p, int gap)
{
    v8h lo = *(const v8h*)p;
    v8h hi = *(const v8h*)(p + gap);
    return __builtin_shufflevector(lo, hi,
           0,1,2,3,4,5,6,7,8,9,10,11,12,13,14,15);
}

// =====================================================================
// WMMA GEMM (NT): C[M,N] = A[M,K] (f16, row-major) * B[N,K]^T (f16, row-major)
// f32 accumulate. mode: 0 = store, 1 = C += result, 2 = nontemporal store.
// One wave (32 threads) per 32x64 output tile (2 A-frags x 4 B-frags,
// 8 WMMAs per 32-wide K step). grid = (ceil(N/64), M/32), block = 32.
// Requirements: M % 32 == 0, K % 32 == 0. N-edge handled by clamping the
// B row pointer (loads stay legal; garbage accumulators are never stored).
// Fragment layouts per CDNA5 ISA 7.12.2 (16-bit A 16x32, B 32x16, f32 C/D).
// =====================================================================
__global__ void __launch_bounds__(32)
wmma_gemm_nt(const _Float16* __restrict__ A, const _Float16* __restrict__ B,
             float* __restrict__ C, int M, int N, int K, int mode)
{
    const int lane = threadIdx.x & 31;
    const int half = lane >> 4;      // 0 or 1
    const int l16  = lane & 15;
    const int kA   = half * 8;       // A K-chunk base per lane half (chunks at +0,+16)
    const int kB   = half * 8;       // B K-chunk base per lane half (chunks at +0,+16)
    const int n0   = blockIdx.x * 64;
    const int m0   = blockIdx.y * 32;

    // A row pointers for the two 16-row sub-tiles (already offset by kA)
    const _Float16* Ap0 = A + (size_t)(m0 + l16)      * K + kA;
    const _Float16* Ap1 = A + (size_t)(m0 + 16 + l16) * K + kA;

    // B row pointers per 16-col sub-tile, clamped to a legal row
    const _Float16* Bp[4];
    #pragma unroll
    for (int s = 0; s < 4; ++s) {
        int n = n0 + s * 16 + l16;
        if (n >= N) n = N - 1;                    // clamp: load legal, never stored
        Bp[s] = B + (size_t)n * K + kB;
    }

    v8f acc0[4] = { {}, {}, {}, {} };
    v8f acc1[4] = { {}, {}, {}, {} };

    for (int k = 0; k < K; k += 32) {
        v16h a0 = load_frag(Ap0 + k, 16);
        v16h a1 = load_frag(Ap1 + k, 16);
        #pragma unroll
        for (int s = 0; s < 4; ++s) {
            v16h b = load_frag(Bp[s] + k, 16);
            acc0[s] = __builtin_amdgcn_wmma_f32_16x16x32_f16(
                          false, a0, false, b, (short)0, acc0[s], false, false);
            acc1[s] = __builtin_amdgcn_wmma_f32_16x16x32_f16(
                          false, a1, false, b, (short)0, acc1[s], false, false);
        }
    }

    // D layout: VGPR r -> row (mTile + 8*half + r), col = l16
    #pragma unroll
    for (int mh = 0; mh < 2; ++mh) {
        const int mBase = m0 + mh * 16 + half * 8;
        #pragma unroll
        for (int s = 0; s < 4; ++s) {
            int n = n0 + s * 16 + l16;
            if (n >= N) continue;
            float* Cp = C + (size_t)mBase * N + n;
            const v8f& acc = mh ? acc1[s] : acc0[s];
            if (mode == 1) {
                #pragma unroll
                for (int r = 0; r < 8; ++r)
                    Cp[(size_t)r * N] += acc[r];
            } else if (mode == 2) {
                #pragma unroll
                for (int r = 0; r < 8; ++r)
                    __builtin_nontemporal_store(acc[r], Cp + (size_t)r * N);
            } else {
                #pragma unroll
                for (int r = 0; r < 8; ++r)
                    Cp[(size_t)r * N] = acc[r];
            }
        }
    }
}

// ---------------- embedding gather ----------------
__global__ void embed_gather(const int* __restrict__ ids,
                             const float* __restrict__ E,
                             float* __restrict__ h)
{
    int t = blockIdx.x;
    int id = ids[t];
    const float* src = E + (size_t)id * D_MODEL;
    float* dst = h + (size_t)t * D_MODEL;
    for (int d = threadIdx.x; d < D_MODEL; d += blockDim.x) dst[d] = src[d];
}

// ---------------- rmsnorm -> f16 ----------------
__global__ void rmsnorm_f16(const float* __restrict__ h,
                            const float* __restrict__ w,
                            _Float16* __restrict__ out)
{
    __shared__ float red[256];
    int t = blockIdx.x;
    const float* hr = h + (size_t)t * D_MODEL;
    float ss = 0.f;
    for (int d = threadIdx.x; d < D_MODEL; d += 256) { float v = hr[d]; ss += v * v; }
    red[threadIdx.x] = ss;
    __syncthreads();
    for (int off = 128; off > 0; off >>= 1) {
        if (threadIdx.x < off) red[threadIdx.x] += red[threadIdx.x + off];
        __syncthreads();
    }
    float inv = rsqrtf(red[0] / (float)D_MODEL + EPS_RMS);
    for (int d = threadIdx.x; d < D_MODEL; d += 256)
        out[(size_t)t * D_MODEL + d] = (_Float16)(hr[d] * inv * w[d]);
}

// ---------------- depthwise conv (k=4, causal) + SiLU ----------------
// xz: [T, 2*D_INNER]; x half is xz[t, 0..D_INNER)
__global__ void conv_silu(const float* __restrict__ xz,
                          const float* __restrict__ cw,
                          const float* __restrict__ cb,
                          float* __restrict__ x32,
                          _Float16* __restrict__ x16)
{
    int idx = blockIdx.x * blockDim.x + threadIdx.x;  // t*D_INNER + c
    int c = idx & (D_INNER - 1);
    int t = idx >> 11;
    if (t >= T_SEQ) return;
    float acc = cb[c];
    #pragma unroll
    for (int k = 0; k < D_CONV; ++k) {
        int tt = t + k - (D_CONV - 1);
        float v = (tt >= 0) ? xz[(size_t)tt * (2 * D_INNER) + c] : 0.f;
        acc += cw[c * D_CONV + k] * v;
    }
    float s = acc / (1.f + __expf(-acc));   // silu
    x32[idx] = s;
    x16[idx] = (_Float16)s;
}

// ---------------- generic f32 -> f16 ----------------
__global__ void cvt_f16(const float* __restrict__ src, _Float16* __restrict__ dst,
                        long long n)
{
    long long i = (long long)blockIdx.x * blockDim.x + threadIdx.x;
    if (i < n) dst[i] = (_Float16)src[i];
}

// ---------------- strided slice f32 -> f16 (x_dbl[:, :DT_RANK]) ----------------
__global__ void slice_cvt_f16(const float* __restrict__ src, _Float16* __restrict__ dst,
                              int rows, int cols, int ld)
{
    int i = blockIdx.x * blockDim.x + threadIdx.x;
    if (i >= rows * cols) return;
    int r = i / cols, c = i - r * cols;
    dst[i] = (_Float16)src[(size_t)r * ld + c];
}

// ---------------- softplus(x + bias) in place ----------------
__global__ void softplus_bias(float* __restrict__ dt, const float* __restrict__ b)
{
    int i = blockIdx.x * blockDim.x + threadIdx.x;
    if (i >= T_SEQ * D_INNER) return;
    int c = i & (D_INNER - 1);
    float v = dt[i] + b[c];
    dt[i] = (v > 20.f) ? v : log1pf(__expf(v));
}

// ---------------- selective scan ----------------
// 16 lanes per channel hold the 16 states; y reduced with shfl_xor.
// grid = D_INNER/16 blocks of 256 threads (16 channels x 16 states).
__global__ void selective_scan(const float* __restrict__ dt,
                               const float* __restrict__ xz,   // for z half
                               const float* __restrict__ x32,  // u
                               const float* __restrict__ xdbl, // B,C at cols 64..95
                               const float* __restrict__ A_log,
                               const float* __restrict__ Dp,
                               _Float16* __restrict__ y16)
{
    const int n    = threadIdx.x & 15;                    // state index
    const int cgrp = threadIdx.x >> 4;                    // channel within block
    const int c    = blockIdx.x * 16 + cgrp;
    const float A  = -__expf(A_log[c * D_STATE + n]);
    const float Dv = Dp[c];
    float s = 0.f;

    for (int t = 0; t < T_SEQ; ++t) {
        float dtv = dt[(size_t)t * D_INNER + c];
        float u   = x32[(size_t)t * D_INNER + c];
        float Bv  = xdbl[(size_t)t * 96 + DT_RANK + n];
        float Cv  = xdbl[(size_t)t * 96 + DT_RANK + D_STATE + n];
        float dA  = __expf(dtv * A);
        s = s * dA + dtv * u * Bv;
        float part = s * Cv;
        part += __shfl_xor(part, 1, 32);
        part += __shfl_xor(part, 2, 32);
        part += __shfl_xor(part, 4, 32);
        part += __shfl_xor(part, 8, 32);
        if (n == 0) {
            float z = xz[(size_t)t * (2 * D_INNER) + D_INNER + c];
            float y = part + u * Dv;
            y *= z / (1.f + __expf(-z));                  // y * silu(z)
            y16[(size_t)t * D_INNER + c] = (_Float16)y;
        }
    }
}

// =====================================================================
static inline int cdiv(int a, int b) { return (a + b - 1) / b; }

extern "C" void kernel_launch(void* const* d_in, const int* in_sizes, int n_in,
                              void* d_out, int out_size, void* d_ws, size_t ws_size,
                              hipStream_t stream)
{
    const int*   ids      = (const int*)  d_in[0];
    const float* E        = (const float*)d_in[1];
    const float* in_proj  = (const float*)d_in[2];   // [2,4096,1024]
    const float* conv_w   = (const float*)d_in[3];   // [2,2048,4]
    const float* conv_b   = (const float*)d_in[4];   // [2,2048]
    const float* x_proj   = (const float*)d_in[5];   // [2,96,2048]
    const float* dt_proj  = (const float*)d_in[6];   // [2,2048,64]
    const float* dt_projb = (const float*)d_in[7];   // [2,2048]
    const float* A_log    = (const float*)d_in[8];   // [2,2048,16]
    const float* Dparam   = (const float*)d_in[9];   // [2,2048]
    const float* out_proj = (const float*)d_in[10];  // [2,1024,2048]
    const float* norm_w   = (const float*)d_in[11];  // [2,1024]
    const float* norm_f   = (const float*)d_in[12];  // [1024]
    float* logits = (float*)d_out;                   // [2048, 50257] f32

    // ---- workspace bump allocator (all offsets multiples of 256B) ----
    char* ws = (char*)d_ws;
    size_t off = 0;
    auto alloc = [&](size_t bytes) -> char* {
        char* p = ws + off;
        off += (bytes + 255) & ~(size_t)255;
        return p;
    };
    float*    h     = (float*)   alloc((size_t)T_SEQ * D_MODEL * 4);        //  8 MB
    _Float16* hn16  = (_Float16*)alloc((size_t)T_SEQ * D_MODEL * 2);        //  4 MB
    float*    xz    = (float*)   alloc((size_t)T_SEQ * 2 * D_INNER * 4);    // 32 MB
    float*    x32   = (float*)   alloc((size_t)T_SEQ * D_INNER * 4);        // 16 MB
    _Float16* x16   = (_Float16*)alloc((size_t)T_SEQ * D_INNER * 2);        //  8 MB
    float*    xdbl  = (float*)   alloc((size_t)T_SEQ * 96 * 4);             //  .75MB
    _Float16* dt16  = (_Float16*)alloc((size_t)T_SEQ * DT_RANK * 2);        //  .25MB
    float*    dtln  = (float*)   alloc((size_t)T_SEQ * D_INNER * 4);        // 16 MB
    _Float16* y16   = (_Float16*)alloc((size_t)T_SEQ * D_INNER * 2);        //  8 MB
    _Float16* wA16  = (_Float16*)alloc((size_t)2 * D_INNER * D_MODEL * 2);  //  8 MB
    _Float16* wB16  = (_Float16*)alloc((size_t)96 * D_INNER * 2);           //  .4 MB
    _Float16* wC16  = (_Float16*)alloc((size_t)D_INNER * DT_RANK * 2);      //  .25MB
    _Float16* wD16  = (_Float16*)alloc((size_t)D_MODEL * D_INNER * 2);      //  4 MB
    _Float16* E16   = (_Float16*)alloc((size_t)VOCAB * D_MODEL * 2);        // 103 MB
    (void)ws_size;

    // embedding -> f16 (only needed at the end, but independent: start early)
    {
        long long n = (long long)VOCAB * D_MODEL;
        cvt_f16<<<(int)((n + 255) / 256), 256, 0, stream>>>(E, E16, n);
    }

    // h = embedding[ids]
    embed_gather<<<T_SEQ, 256, 0, stream>>>(ids, E, h);

    for (int L = 0; L < N_LAYER; ++L) {
        const float* ipw = in_proj  + (size_t)L * 2 * D_INNER * D_MODEL;
        const float* cw  = conv_w   + (size_t)L * D_INNER * D_CONV;
        const float* cb  = conv_b   + (size_t)L * D_INNER;
        const float* xpw = x_proj   + (size_t)L * 96 * D_INNER;
        const float* dpw = dt_proj  + (size_t)L * D_INNER * DT_RANK;
        const float* dpb = dt_projb + (size_t)L * D_INNER;
        const float* Al  = A_log    + (size_t)L * D_INNER * D_STATE;
        const float* Dl  = Dparam   + (size_t)L * D_INNER;
        const float* opw = out_proj + (size_t)L * D_MODEL * D_INNER;
        const float* nw  = norm_w   + (size_t)L * D_MODEL;

        // hn16 = rmsnorm(h)
        rmsnorm_f16<<<T_SEQ, 256, 0, stream>>>(h, nw, hn16);

        // xz = hn @ in_proj^T   [2048,4096]  K=1024
        {
            long long n = (long long)2 * D_INNER * D_MODEL;
            cvt_f16<<<(int)((n + 255) / 256), 256, 0, stream>>>(ipw, wA16, n);
            dim3 g(cdiv(2 * D_INNER, 64), T_SEQ / 32);
            wmma_gemm_nt<<<g, 32, 0, stream>>>(hn16, wA16, xz,
                                               T_SEQ, 2 * D_INNER, D_MODEL, 0);
        }

        // x = silu(conv1d(x_half(xz)))
        conv_silu<<<(T_SEQ * D_INNER) / 256, 256, 0, stream>>>(xz, cw, cb, x32, x16);

        // x_dbl = x @ x_proj^T  [2048,96]  K=2048
        {
            long long n = (long long)96 * D_INNER;
            cvt_f16<<<(int)((n + 255) / 256), 256, 0, stream>>>(xpw, wB16, n);
            dim3 g(cdiv(96, 64), T_SEQ / 32);
            wmma_gemm_nt<<<g, 32, 0, stream>>>(x16, wB16, xdbl,
                                               T_SEQ, 96, D_INNER, 0);
        }

        // dt = softplus(x_dbl[:, :64] @ dt_proj^T + b)  [2048,2048] K=64
        slice_cvt_f16<<<(T_SEQ * DT_RANK) / 256, 256, 0, stream>>>(
            xdbl, dt16, T_SEQ, DT_RANK, 96);
        {
            long long n = (long long)D_INNER * DT_RANK;
            cvt_f16<<<(int)((n + 255) / 256), 256, 0, stream>>>(dpw, wC16, n);
            dim3 g(cdiv(D_INNER, 64), T_SEQ / 32);
            wmma_gemm_nt<<<g, 32, 0, stream>>>(dt16, wC16, dtln,
                                               T_SEQ, D_INNER, DT_RANK, 0);
        }
        softplus_bias<<<(T_SEQ * D_INNER) / 256, 256, 0, stream>>>(dtln, dpb);

        // sequential state scan -> y16 (includes +x*D and *silu(z))
        selective_scan<<<D_INNER / 16, 256, 0, stream>>>(
            dtln, xz, x32, xdbl, Al, Dl, y16);

        // h += y @ out_proj^T  [2048,1024] K=2048
        {
            long long n = (long long)D_MODEL * D_INNER;
            cvt_f16<<<(int)((n + 255) / 256), 256, 0, stream>>>(opw, wD16, n);
            dim3 g(cdiv(D_MODEL, 64), T_SEQ / 32);
            wmma_gemm_nt<<<g, 32, 0, stream>>>(y16, wD16, h,
                                               T_SEQ, D_MODEL, D_INNER, 1);
        }
    }

    // final rmsnorm, then logits = hn @ E^T  [2048,50257] K=1024
    // (nontemporal store: 412 MB streamed out once, keep E16 resident in L2)
    rmsnorm_f16<<<T_SEQ, 256, 0, stream>>>(h, norm_f, hn16);
    {
        dim3 g(cdiv(VOCAB, 64), T_SEQ / 32);
        wmma_gemm_nt<<<g, 32, 0, stream>>>(hn16, E16, logits,
                                           T_SEQ, VOCAB, D_MODEL, 2);
    }
}
